// Encoder_35364760715946
// MI455X (gfx1250) — compile-verified, matrix operand order checked by reading, not run on hardware
//
#include <hip/hip_runtime.h>

// ---------------------------------------------------------------------------
// MI455X (gfx1250) implementation:
//   3x [Conv1D(K=5,SAME,512->512) + BN(inference) + ReLU]  -> bi-LSTM(U=256)
// All GEMM work runs on v_wmma_f32_16x16x32_f16 (f16 in, f32 accumulate).
//
// Conv: compute-bound GEMM (M=65536,N=512,K=2560) -> 128x128 block tile,
//       8 waves x (2x4) 16x16 WMMA tiles, double-buffered LDS staging.
// LSTM: persistent kernel, 2 WGs (fw/bw), 32 waves; per step fuses
//       x_t@Wk (K=512) + h@Wr (K=256) into one 64x1024 WMMA accumulation;
//       weight chunks staged through LDS once per chunk (shared by all
//       m-tile waves); cell state c lives in VGPRs for all 1024 steps.
// ---------------------------------------------------------------------------

typedef __attribute__((ext_vector_type(16))) _Float16 v16h;
typedef __attribute__((ext_vector_type(8)))  float    v8f;

union FragH { v16h h; unsigned int u[8]; };

#define B_   64
#define T_   1024
#define D_   512
#define K_   5
#define NC_  3
#define U_   256
#define G_   1024   /* 4*U */
#define KR_  2560   /* K_*D_  (im2col reduction length) */

// WMMA 16-bit A/B fragment dword index inside a 32-K-chunk row:
//   lane = (half<<4)|lr ; VGPR v holds K-pair starting at
//   kbase = (v>>2)*16 + half*8 + (v&3)*2  -> dword = (v>>2)*8 + half*4 + (v&3)
// For fixed half this is dwords {0..3}/{8..11} (+half*4) -> lowers to 2x ds_load_b128.

// ---------------------------------------------------------------------------
// Packing / conversion kernels
// ---------------------------------------------------------------------------
__global__ void cvt_f16(const float* __restrict__ in, _Float16* __restrict__ outp, int n) {
    int i = blockIdx.x * blockDim.x + threadIdx.x;
    if (i < n) outp[i] = (_Float16)in[i];
}

// conv_kernels [NC][K][D][D] (k,d,n)  ->  Bt[NC][n=512][r=K*512] f16
__global__ void pack_conv(const float* __restrict__ w, _Float16* __restrict__ outp) {
    int idx = blockIdx.x * blockDim.x + threadIdx.x;
    if (idx >= NC_ * D_ * KR_) return;
    int i   = idx / (D_ * KR_);
    int rem = idx - i * (D_ * KR_);
    int n   = rem / KR_;
    int r   = rem - n * KR_;
    int k   = r >> 9;        // r / 512
    int d   = r & 511;
    outp[idx] = (_Float16)w[(((size_t)i * K_ + k) * D_ + d) * D_ + n];
}

// src [srcRows][srcCols] f32 -> dst [srcCols][srcRows] f16   (weight transpose)
__global__ void pack_T(const float* __restrict__ w, _Float16* __restrict__ outp,
                       int srcRows, int srcCols) {
    int idx = blockIdx.x * blockDim.x + threadIdx.x;
    if (idx >= srcRows * srcCols) return;
    int g = idx / srcRows;   // dst row  = src col
    int d = idx - g * srcRows;
    outp[idx] = (_Float16)w[(size_t)d * srcCols + g];
}

// ---------------------------------------------------------------------------
// Conv1D(SAME,K=5) + BN + ReLU as a WMMA GEMM over r = tap*512 + d.
// Block: 256 threads (8 waves). Output tile 128(M) x 128(N).
//   wave w: m-quadrant wy=w&3 (2 mtiles), n-half wx=w>>2 (4 ntiles).
// Double-buffered LDS (one barrier per K-chunk); global loads for chunk
// kc+1 are issued before the 8 WMMAs of chunk kc.
// ---------------------------------------------------------------------------
__global__ __launch_bounds__(256)
void conv_bn_relu(const _Float16* __restrict__ xin,   // [B,T,D] f16
                  const _Float16* __restrict__ Bt,    // [512][2560] f16
                  const float* __restrict__ bias,
                  const float* __restrict__ gamma,
                  const float* __restrict__ beta,
                  const float* __restrict__ mean,
                  const float* __restrict__ var,
                  _Float16* __restrict__ xout)        // [B,T,D] f16
{
    __shared__ unsigned int lsA[2][128 * 16];   // 2 x 8 KB
    __shared__ unsigned int lsB[2][128 * 16];   // 2 x 8 KB

    const int mb   = blockIdx.x >> 2;
    const int nb   = blockIdx.x & 3;
    const int m0   = mb << 7;
    const int n0   = nb << 7;
    const int t0   = m0 & (T_ - 1);
    const int bidx = m0 >> 10;

    const int tid  = threadIdx.x;
    const int lane = tid & 31;
    const int wave = tid >> 5;
    const int wy   = wave & 3;            // -> mtiles wy*2 + {0,1}
    const int wx   = wave >> 2;           // -> ntiles wx*4 + {0..3}
    const int half = lane >> 4;
    const int lr   = lane & 15;

    const int lrow  = tid >> 1;           // 0..127 (staging row)
    const int lhalf = tid & 1;            // dwords lhalf*8 .. +7

    v8f acc[2][4];
    #pragma unroll
    for (int my = 0; my < 2; ++my)
        #pragma unroll
        for (int nx = 0; nx < 4; ++nx) acc[my][nx] = (v8f){};

    uint4 ra0, ra1, rb0, rb1;

    auto loadRegs = [&](int kk) {
        const int tap   = kk >> 4;
        const int dbase = (kk & 15) << 5;
        int tt = t0 + lrow + tap - 2;
        ra0 = make_uint4(0u, 0u, 0u, 0u);
        ra1 = ra0;
        if (tt >= 0 && tt < T_) {
            const _Float16* p = xin +
                ((size_t)(bidx * T_ + tt) * D_ + dbase + lhalf * 16);
            ra0 = *(const uint4*)p;
            ra1 = *(const uint4*)(p + 8);
        }
        const _Float16* q = Bt +
            ((size_t)(n0 + lrow) * KR_ + (kk << 5) + lhalf * 16);
        rb0 = *(const uint4*)q;
        rb1 = *(const uint4*)(q + 8);
    };
    auto storeRegs = [&](int buf) {
        uint4* da = (uint4*)(lsA[buf] + lrow * 16 + lhalf * 8);
        da[0] = ra0; da[1] = ra1;
        uint4* db = (uint4*)(lsB[buf] + lrow * 16 + lhalf * 8);
        db[0] = rb0; db[1] = rb1;
    };

    loadRegs(0);
    storeRegs(0);
    __syncthreads();
    int p = 0;

    for (int kc = 0; kc < 80; ++kc) {
        if (kc + 1 < 80) loadRegs(kc + 1);              // overlap with WMMA
        if (kc + 4 < 80)                                // L2 prefetch ahead
            __builtin_prefetch(Bt + ((size_t)(n0 + lrow) * KR_ + ((kc + 4) << 5)), 0, 0);

        FragH a[2], b[4];
        #pragma unroll
        for (int my = 0; my < 2; ++my) {
            const unsigned int* base = lsA[p] + ((wy * 2 + my) * 16 + lr) * 16;
            #pragma unroll
            for (int v = 0; v < 8; ++v)
                a[my].u[v] = base[(v >> 2) * 8 + half * 4 + (v & 3)];
        }
        #pragma unroll
        for (int nx = 0; nx < 4; ++nx) {
            const unsigned int* base = lsB[p] + ((wx * 4 + nx) * 16 + lr) * 16;
            #pragma unroll
            for (int v = 0; v < 8; ++v)
                b[nx].u[v] = base[(v >> 2) * 8 + half * 4 + (v & 3)];
        }
        #pragma unroll
        for (int my = 0; my < 2; ++my)
            #pragma unroll
            for (int nx = 0; nx < 4; ++nx)
                acc[my][nx] = __builtin_amdgcn_wmma_f32_16x16x32_f16(
                    false, a[my].h, false, b[nx].h, (short)0, acc[my][nx], false, false);

        if (kc + 1 < 80) {
            storeRegs(p ^ 1);     // write the *other* buffer: no read conflict
            __syncthreads();
            p ^= 1;
        }
    }

    // ---- epilogue: bias + BN(inference) + ReLU -> f16 ----
    #pragma unroll
    for (int nx = 0; nx < 4; ++nx) {
        const int n = n0 + (wx * 4 + nx) * 16 + lr;
        const float s  = gamma[n] * rsqrtf(var[n] + 1e-3f);
        const float sh = beta[n] - mean[n] * s;
        const float bi = bias[n];
        #pragma unroll
        for (int my = 0; my < 2; ++my) {
            #pragma unroll
            for (int r = 0; r < 8; ++r) {
                const int m = m0 + (wy * 2 + my) * 16 + r + half * 8;
                float v = (acc[my][nx][r] + bi) * s + sh;
                v = v > 0.f ? v : 0.f;
                xout[(size_t)m * D_ + n] = (_Float16)v;
            }
        }
    }
}

// ---------------------------------------------------------------------------
// Persistent bidirectional LSTM. grid = 2 (dir), block = 1024 (32 waves).
// Dynamic LDS: hA (32 KB, persistent h f16) + xtile (4 KB) + bT (64 KB,
// cooperatively staged 1024x32 weight chunk, shared by all m-tile waves).
// Wave w owns mtile = w&3 and, for each gate q, ntiles q*16 + (w>>2)*2+{0,1}
// => i/f/g/o for the same 32 units live in one wave; c stays in VGPRs.
// ---------------------------------------------------------------------------
__global__ __launch_bounds__(1024)
void bilstm(const _Float16* __restrict__ x16,     // [B,T,D] f16 (conv output)
            const _Float16* __restrict__ WkT_fw,  // [1024][512] f16
            const _Float16* __restrict__ WkT_bw,
            const _Float16* __restrict__ WrT_fw,  // [1024][256] f16
            const _Float16* __restrict__ WrT_bw,
            const float* __restrict__ bias_fw,    // [1024]
            const float* __restrict__ bias_bw,
            float* __restrict__ out)              // [B,T,512] f32
{
    extern __shared__ char smem[];
    _Float16*     hA    = (_Float16*)smem;                        // 32 KB
    unsigned int* xtile = (unsigned int*)(smem + 32 * 1024);      //  4 KB
    unsigned int* bT    = (unsigned int*)(smem + 36 * 1024);      // 64 KB

    const int dir = blockIdx.x;
    const unsigned int* Wku  = (const unsigned int*)(dir ? WkT_bw : WkT_fw); // row stride 256 dw
    const unsigned int* Wru  = (const unsigned int*)(dir ? WrT_bw : WrT_fw); // row stride 128 dw
    const float*        bias = dir ? bias_bw : bias_fw;
    const unsigned int* xu   = (const unsigned int*)x16;

    const int tid  = threadIdx.x;
    const int lane = tid & 31;
    const int wave = tid >> 5;     // 0..31
    const int mt   = wave & 3;
    const int wg   = wave >> 2;    // 0..7
    const int half = lane >> 4;
    const int lr   = lane & 15;

    // zero h state
    {
        unsigned int* hz = (unsigned int*)hA;
        #pragma unroll
        for (int i = 0; i < 8; ++i) hz[tid + i * 1024] = 0u;
    }
    v8f c0 = {}; v8f c1 = {};
    __syncthreads();

    for (int t = 0; t < T_; ++t) {
        const int tc = dir ? (T_ - 1 - t) : t;

        // init accumulators with gate bias (broadcast along rows)
        v8f acc[8];
        #pragma unroll
        for (int q = 0; q < 4; ++q)
            #pragma unroll
            for (int s = 0; s < 2; ++s) {
                const float bv = bias[q * 256 + (wg * 2 + s) * 16 + lr];
                v8f z;
                #pragma unroll
                for (int r = 0; r < 8; ++r) z[r] = bv;
                acc[q * 2 + s] = z;
            }

        // ---- x_t @ Wk : K = 512, 16 chunks; x tile + weight chunk in LDS ----
        for (int kc = 0; kc < 16; ++kc) {
            {   // stage x_t chunk [64 rows x 16 dwords]
                const int row = tid >> 4;
                const int c   = tid & 15;
                xtile[row * 16 + c] =
                    xu[(size_t)(row * T_ + tc) * 256 + (kc << 4) + c];
            }
            #pragma unroll
            for (int i = 0; i < 4; ++i) {   // stage Wk chunk [1024 x 16 dwords]
                const int flat = i * 4096 + (tid << 2);
                const int row  = flat >> 4;
                const int c    = flat & 15;
                *(uint4*)(bT + flat) =
                    *(const uint4*)(Wku + (size_t)row * 256 + (kc << 4) + c);
            }
            if (kc + 1 < 16)   // prefetch next chunk row (one row per thread)
                __builtin_prefetch(Wku + (size_t)tid * 256 + ((kc + 1) << 4), 0, 0);
            __syncthreads();

            FragH a;
            {
                const unsigned int* base = xtile + (mt * 16 + lr) * 16;
                #pragma unroll
                for (int v = 0; v < 8; ++v)
                    a.u[v] = base[(v >> 2) * 8 + half * 4 + (v & 3)];
            }
            #pragma unroll
            for (int q = 0; q < 4; ++q)
                #pragma unroll
                for (int s = 0; s < 2; ++s) {
                    const int n = q * 256 + (wg * 2 + s) * 16 + lr;
                    FragH b;
                    const unsigned int* bb = bT + n * 16;
                    #pragma unroll
                    for (int v = 0; v < 8; ++v)
                        b.u[v] = bb[(v >> 2) * 8 + half * 4 + (v & 3)];
                    acc[q * 2 + s] = __builtin_amdgcn_wmma_f32_16x16x32_f16(
                        false, a.h, false, b.h, (short)0, acc[q * 2 + s], false, false);
                }
            __syncthreads();
        }

        // ---- h @ Wr : K = 256, 8 chunks; h from persistent LDS ----
        for (int kc = 0; kc < 8; ++kc) {
            #pragma unroll
            for (int i = 0; i < 4; ++i) {   // stage Wr chunk [1024 x 16 dwords]
                const int flat = i * 4096 + (tid << 2);
                const int row  = flat >> 4;
                const int c    = flat & 15;
                *(uint4*)(bT + flat) =
                    *(const uint4*)(Wru + (size_t)row * 128 + (kc << 4) + c);
            }
            if (kc + 1 < 8)
                __builtin_prefetch(Wru + (size_t)tid * 128 + ((kc + 1) << 4), 0, 0);
            __syncthreads();

            FragH a;
            {
                const unsigned int* base =
                    (const unsigned int*)hA + (mt * 16 + lr) * 128 + (kc << 4);
                #pragma unroll
                for (int v = 0; v < 8; ++v)
                    a.u[v] = base[(v >> 2) * 8 + half * 4 + (v & 3)];
            }
            #pragma unroll
            for (int q = 0; q < 4; ++q)
                #pragma unroll
                for (int s = 0; s < 2; ++s) {
                    const int n = q * 256 + (wg * 2 + s) * 16 + lr;
                    FragH b;
                    const unsigned int* bb = bT + n * 16;
                    #pragma unroll
                    for (int v = 0; v < 8; ++v)
                        b.u[v] = bb[(v >> 2) * 8 + half * 4 + (v & 3)];
                    acc[q * 2 + s] = __builtin_amdgcn_wmma_f32_16x16x32_f16(
                        false, a.h, false, b.h, (short)0, acc[q * 2 + s], false, false);
                }
            __syncthreads();   // also guarantees hA reads done on last chunk
        }

        // ---- fused LSTM cell update (keras gate order i,f,g,o) ----
        #pragma unroll
        for (int s = 0; s < 2; ++s) {
            v8f* cp = s ? &c1 : &c0;
            const int u = (wg * 2 + s) * 16 + lr;
            #pragma unroll
            for (int r = 0; r < 8; ++r) {
                const float zi = acc[0 + s][r];
                const float zf = acc[2 + s][r];
                const float zg = acc[4 + s][r];
                const float zo = acc[6 + s][r];
                const float ig = 1.f / (1.f + __expf(-zi));
                const float fg = 1.f / (1.f + __expf(-zf));
                const float gg = tanhf(zg);
                const float og = 1.f / (1.f + __expf(-zo));
                const float cv = fg * (*cp)[r] + ig * gg;
                (*cp)[r] = cv;
                const float hv = og * tanhf(cv);
                const int row = mt * 16 + r + half * 8;      // batch index
                hA[row * 256 + u] = (_Float16)hv;
                out[((size_t)row * T_ + tc) * 512 + dir * 256 + u] = hv;
            }
        }
        __syncthreads();
    }
}

// ---------------------------------------------------------------------------
// Host launcher
// ---------------------------------------------------------------------------
extern "C" void kernel_launch(void* const* d_in, const int* in_sizes, int n_in,
                              void* d_out, int out_size, void* d_ws, size_t ws_size,
                              hipStream_t stream) {
    (void)in_sizes; (void)n_in; (void)out_size; (void)ws_size;

    const float* x   = (const float*)d_in[0];
    const float* cw  = (const float*)d_in[1];
    const float* cb  = (const float*)d_in[2];
    const float* gm  = (const float*)d_in[3];
    const float* bt  = (const float*)d_in[4];
    const float* mn  = (const float*)d_in[5];
    const float* vr  = (const float*)d_in[6];
    const float* kfw = (const float*)d_in[7];
    const float* rfw = (const float*)d_in[8];
    const float* bfw = (const float*)d_in[9];
    const float* kbw = (const float*)d_in[10];
    const float* rbw = (const float*)d_in[11];
    const float* bbw = (const float*)d_in[12];
    float* outp = (float*)d_out;

    char* ws = (char*)d_ws;
    const size_t xbytes = (size_t)B_ * T_ * D_ * 2;       // 64 MB
    _Float16* ping = (_Float16*)ws;
    _Float16* pong = (_Float16*)(ws + xbytes);
    _Float16* btc  = (_Float16*)(ws + 2 * xbytes);                         // 7.5 MB
    _Float16* wktf = (_Float16*)(ws + 2 * xbytes + (size_t)NC_ * D_ * KR_ * 2);
    _Float16* wktb = wktf + (size_t)G_ * D_;
    _Float16* wrtf = wktb + (size_t)G_ * D_;
    _Float16* wrtb = wrtf + (size_t)G_ * U_;

    const int nX = B_ * T_ * D_;
    cvt_f16  <<<(nX + 255) / 256, 256, 0, stream>>>(x, ping, nX);
    pack_conv<<<(NC_ * D_ * KR_ + 255) / 256, 256, 0, stream>>>(cw, btc);
    pack_T   <<<(G_ * D_ + 255) / 256, 256, 0, stream>>>(kfw, wktf, D_, G_);
    pack_T   <<<(G_ * D_ + 255) / 256, 256, 0, stream>>>(kbw, wktb, D_, G_);
    pack_T   <<<(G_ * U_ + 255) / 256, 256, 0, stream>>>(rfw, wrtf, U_, G_);
    pack_T   <<<(G_ * U_ + 255) / 256, 256, 0, stream>>>(rbw, wrtb, U_, G_);

    const dim3 cgrid(2048);   // (M/128)=512 x (N/128)=4
    conv_bn_relu<<<cgrid, 256, 0, stream>>>(ping, btc,
        cb, gm, bt, mn, vr, pong);
    conv_bn_relu<<<cgrid, 256, 0, stream>>>(pong, btc + (size_t)D_ * KR_,
        cb + D_, gm + D_, bt + D_, mn + D_, vr + D_, ping);
    conv_bn_relu<<<cgrid, 256, 0, stream>>>(ping, btc + (size_t)2 * D_ * KR_,
        cb + 2 * D_, gm + 2 * D_, bt + 2 * D_, mn + 2 * D_, vr + 2 * D_, pong);

    bilstm<<<2, 1024, 100 * 1024, stream>>>(pong, wktf, wktb, wrtf, wrtb,
                                            bfw, bbw, outp);
}
